// EHRMemoryNetwork_13769665151412
// MI455X (gfx1250) — matrix operand order repaired and checked by reading, not compiled, not run on hardware
//
#include <hip/hip_runtime.h>
#include <hip/hip_bf16.h>

// ---------------------------------------------------------------------------
// EHR memory network for MI455X (gfx1250, wave32, WMMA).
// Sizes: B=32 T=64 MOD=3 D=4 WD=256 MEM=256 HID=512 DEMO=64 N_NODES=4096
// Rows R = B*T*MOD*D = 24576, events = 6144, S = 768 steps/batch.
// ---------------------------------------------------------------------------

typedef __attribute__((ext_vector_type(16))) __bf16 v16bf;
typedef __attribute__((ext_vector_type(8)))  float  v8f;

#define N_NODES 4096
#define Bb 32
#define Tt 64
#define MODm 3
#define Dd 4
#define WD 256
#define MEM 256
#define HID 512
#define DEMO 64
#define ROWS (Bb*Tt*MODm*Dd)   // 24576
#define EVENTS (Bb*Tt*MODm)    // 6144
#define SSTEPS (Tt*MODm*Dd)    // 768

static __device__ __forceinline__ v8f wmma_bf16(v16bf a, v16bf b, v8f c) {
  // D = A(16x32 bf16) x B(32x16 bf16) + C(16x16 f32)
  return __builtin_amdgcn_wmma_f32_16x16x32_bf16(false, a, false, b,
                                                 (short)0, c, false, false);
}

static __device__ __forceinline__ void split2(float v, __bf16 &hi, __bf16 &lo) {
  hi = (__bf16)v;
  lo = (__bf16)(v - (float)hi);
}

static __device__ __forceinline__ unsigned packbf(__bf16 a, __bf16 b) {
  // a -> [15:0] (even K), b -> [31:16] (odd K)
  return ((unsigned)__builtin_bit_cast(unsigned short, b) << 16) |
         (unsigned)__builtin_bit_cast(unsigned short, a);
}

// ---------------------------------------------------------------------------
// K0: pack We/Wa (f32 [K=256, N=256]) into bf16 hi/lo B-matrix fragments.
// Fragment layout (16x16x32 bf16 B-matrix, wave32):
//   lane L: N = ct*16 + (L&15), halfgroup h = L>>4
//   dword i (0..7): K = kk*32 + h*16 + 2i (even K in [15:0], odd in [31:16])
// dword offset(m, p, kk, ct, lane, i) = ((((m*2+p)*8+kk)*16+ct)*32+lane)*8 + i
// ---------------------------------------------------------------------------
__global__ __launch_bounds__(256) void prep_weights(
    const float* __restrict__ We, const float* __restrict__ Wa,
    unsigned* __restrict__ wpack) {
  int id = blockIdx.x * blockDim.x + threadIdx.x;     // 65536 pair-slots
  if (id >= (2 * 8 * 16 * 32 * 8)) return;
  int i    =  id        & 7;
  int lane = (id >> 3)  & 31;
  int ct   = (id >> 8)  & 15;
  int kk   = (id >> 12) & 7;
  int m    = (id >> 15) & 1;
  const float* W = m ? Wa : We;
  int n = ct * 16 + (lane & 15);
  int h = lane >> 4;
  int k = kk * 32 + h * 16 + 2 * i;
  float w0 = W[k * MEM + n];
  float w1 = W[(k + 1) * MEM + n];
  __bf16 h0, l0, h1, l1;
  split2(w0, h0, l0);
  split2(w1, h1, l1);
  unsigned offh = ((((unsigned)(m * 2 + 0) * 8 + kk) * 16 + ct) * 32 + lane) * 8 + i;
  unsigned offl = ((((unsigned)(m * 2 + 1) * 8 + kk) * 16 + ct) * 32 + lane) * 8 + i;
  wpack[offh] = packbf(h0, h1);
  wpack[offl] = packbf(l0, l1);
}

// ---------------------------------------------------------------------------
// K1: demographics residual MLP -> M[b, slot 0, :].  One block per batch.
// ---------------------------------------------------------------------------
__global__ __launch_bounds__(256) void demo_kernel(
    const float* __restrict__ demo,
    const float* __restrict__ W1, const float* __restrict__ b1,
    const float* __restrict__ W2, const float* __restrict__ b2,
    const float* __restrict__ W3, const float* __restrict__ b3,
    float* __restrict__ M) {
  __shared__ float sdemo[DEMO];
  __shared__ float sh[HID];
  __shared__ float sy[DEMO];
  const int b = blockIdx.x;
  const int t = threadIdx.x;
  if (t < DEMO) sdemo[t] = demo[b * DEMO + t];
  __syncthreads();
  // h = relu(demo @ W1 + b1): each thread does 2 of 512 columns
  #pragma unroll
  for (int cc = 0; cc < 2; ++cc) {
    int c = t + cc * 256;
    float acc = b1[c];
    #pragma unroll 8
    for (int k = 0; k < DEMO; ++k) acc += sdemo[k] * W1[k * HID + c];
    sh[c] = acc > 0.f ? acc : 0.f;
  }
  __syncthreads();
  // y = h @ W2 + b2 + demo
  if (t < DEMO) {
    float acc = b2[t];
    #pragma unroll 8
    for (int k = 0; k < HID; ++k) acc += sh[k] * W2[k * DEMO + t];
    sy[t] = acc + sdemo[t];
  }
  __syncthreads();
  // out = y @ W3 + b3 -> M[b, 0, t]
  float acc = b3[t];
  #pragma unroll 8
  for (int k = 0; k < DEMO; ++k) acc += sy[k] * W3[k * MEM + t];
  M[(long)b * N_NODES * MEM + t] = acc;
}

// ---------------------------------------------------------------------------
// K2: broadcast init_mem into slots 1..4095 for every batch (128 MiB stream).
// grid = (1024, 32), block = 256.  slot = 1 + bx*4 + tid/64, j4 = tid%64.
// ---------------------------------------------------------------------------
__global__ __launch_bounds__(256) void init_kernel(
    const float4* __restrict__ im, float4* __restrict__ M4) {
  const int b    = blockIdx.y;
  const int slot = 1 + blockIdx.x * 4 + (threadIdx.x >> 6);
  const int j4   = threadIdx.x & 63;
  if (slot >= N_NODES) return;
  M4[((long)b * N_NODES + slot) * 64 + j4] = im[j4];
}

// ---------------------------------------------------------------------------
// K3: gate GEMMs (bf16 hi/lo split WMMA) + sigmoid/tanh + D-chain compose +
// validity mask -> Af/Bf [B*S, MEM] in workspace.
// Block = 8 waves = 2 row-tiles x 4 col-tile-groups.  The block's 32 x-rows
// are cooperatively converted ONCE into fragment-ordered bf16 hi/lo pairs in
// LDS (32 KB of the WGP's 320 KB), so each wave's A operand is two
// ds_load_b128 per fragment instead of redundant global gathers + converts.
// ---------------------------------------------------------------------------
__global__ __launch_bounds__(256) void gates_kernel(
    const float* __restrict__ x,        // [ROWS, WD]
    const int* __restrict__ mask,       // [B*T]
    const int* __restrict__ vmod,       // [EVENTS]
    const float* __restrict__ be, const float* __restrict__ ba,
    const unsigned* __restrict__ wpack,
    float* __restrict__ Af, float* __restrict__ Bf) {
  // apack dword offset(rtl, p, kk, lane, i) = (((rtl*2+p)*8+kk)*32+lane)*8 + i
  __shared__ unsigned apack[2 * 2 * 8 * 32 * 8];   // 32 KB

  const int brow = blockIdx.x * 32;                // first of 32 rows
  // ---- cooperative A-fragment staging: 4096 pair-slots / 256 threads ----
  #pragma unroll
  for (int it = 0; it < 16; ++it) {
    int s    = it * 256 + threadIdx.x;             // 0..4095
    int i    =  s        & 7;
    int lane = (s >> 3)  & 31;
    int kk   = (s >> 8)  & 7;
    int rtl  = (s >> 11) & 1;
    int hh   = lane >> 4;
    int row  = brow + rtl * 16 + (lane & 15);
    int k    = kk * 32 + hh * 16 + i * 2;
    float2 v = *(const float2*)(x + (long)row * WD + k);
    __bf16 h0, l0, h1, l1;
    split2(v.x, h0, l0);
    split2(v.y, h1, l1);
    apack[(((rtl * 2 + 0) * 8 + kk) * 32 + lane) * 8 + i] = packbf(h0, h1);
    apack[(((rtl * 2 + 1) * 8 + kk) * 32 + lane) * 8 + i] = packbf(l0, l1);
  }
  __syncthreads();

  const int wid  = threadIdx.x >> 5;   // 0..7
  const int lane = threadIdx.x & 31;
  const int rtl  = wid >> 2;           // 0..1 : which 16-row tile
  const int ctg  = wid & 3;            // 0..3 : which 4x16-col group
  const int mrow = lane & 15;
  const int h    = lane >> 4;

  v8f accE[4], accA[4];
  #pragma unroll
  for (int c = 0; c < 4; ++c)
    #pragma unroll
    for (int r = 0; r < 8; ++r) { accE[c][r] = 0.f; accA[c][r] = 0.f; }

  for (int kk = 0; kk < 8; ++kk) {
    v16bf ahi = *(const v16bf*)&apack[(((rtl * 2 + 0) * 8 + kk) * 32 + lane) * 8];
    v16bf alo = *(const v16bf*)&apack[(((rtl * 2 + 1) * 8 + kk) * 32 + lane) * 8];
    #pragma unroll
    for (int c = 0; c < 4; ++c) {
      int ct = ctg * 4 + c;
      // We (mat 0): hi*hi + hi*lo + lo*hi  (~f32 accuracy)
      {
        unsigned offh = (((0u * 8 + kk) * 16 + ct) * 32 + lane) * 8;
        unsigned offl = (((1u * 8 + kk) * 16 + ct) * 32 + lane) * 8;
        v16bf bh = *(const v16bf*)(wpack + offh);
        v16bf bl = *(const v16bf*)(wpack + offl);
        accE[c] = wmma_bf16(ahi, bh, accE[c]);
        accE[c] = wmma_bf16(ahi, bl, accE[c]);
        accE[c] = wmma_bf16(alo, bh, accE[c]);
      }
      // Wa (mat 1)
      {
        unsigned offh = (((2u * 8 + kk) * 16 + ct) * 32 + lane) * 8;
        unsigned offl = (((3u * 8 + kk) * 16 + ct) * 32 + lane) * 8;
        v16bf bh = *(const v16bf*)(wpack + offh);
        v16bf bl = *(const v16bf*)(wpack + offl);
        accA[c] = wmma_bf16(ahi, bh, accA[c]);
        accA[c] = wmma_bf16(ahi, bl, accA[c]);
        accA[c] = wmma_bf16(alo, bh, accA[c]);
      }
    }
  }

  // This lane's 8 C rows are M = h*8 + r  ->  2 whole events (4 levels each).
  const int evbase = blockIdx.x * 8 + rtl * 4 + h * 2;
  int valid2[2];
  #pragma unroll
  for (int e = 0; e < 2; ++e) {
    int ev  = evbase + e;
    int b   = ev / (Tt * MODm);
    int rem = ev % (Tt * MODm);
    int t   = rem / MODm;
    valid2[e] = (mask[b * Tt + t] > 0) && (vmod[ev] > 0);
  }

  #pragma unroll
  for (int c = 0; c < 4; ++c) {
    int n = ctg * 64 + c * 16 + mrow;
    float beb = be[n], bab = ba[n];
    float Ev[8], Av[8];
    #pragma unroll
    for (int r = 0; r < 8; ++r) {
      Ev[r] = 1.f / (1.f + __expf(-(accE[c][r] + beb)));
      Av[r] = tanhf(accA[c][r] + bab);
    }
    #pragma unroll
    for (int e = 0; e < 2; ++e) {
      bool valid = valid2[e] != 0;
      #pragma unroll
      for (int l = 0; l < 4; ++l) {
        float Al = 1.f, Bl = 0.f, s = 1.f;
        #pragma unroll
        for (int m = l; m < 4; ++m) {
          float keep = 1.f - s * Ev[e * 4 + m];
          Al = Al * keep;
          Bl = Bl * keep + s * Av[e * 4 + m];
          s *= 0.5f;
        }
        if (!valid) { Al = 1.f; Bl = 0.f; }
        int g = brow + rtl * 16 + h * 8 + e * 4 + l;  // row in [B*S]
        Af[(long)g * MEM + n] = Al;                   // lanes 0..15 contiguous in n
        Bf[(long)g * MEM + n] = Bl;
      }
    }
  }
}

// ---------------------------------------------------------------------------
// K4: ordered scatter-scan, windowed by 8 to hide L2 RMW latency.
// One block per batch, thread j owns channel j.  Intra-window duplicate ids
// are chained in registers; stores issue in order so the last write wins.
// Same-lane store->load same-address ordering is guaranteed by the ISA.
// Af/Bf are read exactly once -> non-temporal, and the next window is
// prefetched (global_prefetch_b8) to overlap with the dependent M chain.
// ---------------------------------------------------------------------------
__global__ __launch_bounds__(256) void scan_kernel(
    const int* __restrict__ ids,        // [B*S]
    const float* __restrict__ Af, const float* __restrict__ Bf,
    float* __restrict__ M) {
  const int b = blockIdx.x;
  const int j = threadIdx.x;
  const int*   idb = ids + b * SSTEPS;
  const float* af  = Af + (long)b * SSTEPS * MEM + j;
  const float* bf  = Bf + (long)b * SSTEPS * MEM + j;
  float*       Mb  = M + (long)b * N_NODES * MEM + j;

  for (int s0 = 0; s0 < SSTEPS; s0 += 8) {
    if (s0 + 8 < SSTEPS) {                 // pull next window toward L2/L0
      __builtin_prefetch(af + (s0 + 8) * MEM, 0, 1);
      __builtin_prefetch(bf + (s0 + 8) * MEM, 0, 1);
    }
    int idx[8];
    float av[8], bv[8], mv[8], cur[8];
    #pragma unroll
    for (int w = 0; w < 8; ++w) {
      idx[w] = idb[s0 + w];                 // uniform -> SGPR
      av[w]  = __builtin_nontemporal_load(af + (s0 + w) * MEM);
      bv[w]  = __builtin_nontemporal_load(bf + (s0 + w) * MEM);
    }
    #pragma unroll
    for (int w = 0; w < 8; ++w) mv[w] = Mb[idx[w] * MEM];
    #pragma unroll
    for (int w = 0; w < 8; ++w) {
      float base = mv[w];
      #pragma unroll
      for (int v = 0; v < w; ++v)
        if (idx[v] == idx[w]) base = cur[v];  // last prior occurrence wins
      cur[w] = fmaf(base, av[w], bv[w]);
    }
    #pragma unroll
    for (int w = 0; w < 8; ++w) Mb[idx[w] * MEM] = cur[w];  // ordered stores
  }
}

// ---------------------------------------------------------------------------
extern "C" void kernel_launch(void* const* d_in, const int* in_sizes, int n_in,
                              void* d_out, int out_size, void* d_ws, size_t ws_size,
                              hipStream_t stream) {
  const float* x    = (const float*)d_in[0];
  const int*   mask = (const int*)d_in[1];
  const int*   vmod = (const int*)d_in[2];
  const int*   ids  = (const int*)d_in[3];
  const float* demo = (const float*)d_in[4];
  const float* W1 = (const float*)d_in[5];  const float* b1 = (const float*)d_in[6];
  const float* W2 = (const float*)d_in[7];  const float* b2 = (const float*)d_in[8];
  const float* W3 = (const float*)d_in[9];  const float* b3 = (const float*)d_in[10];
  const float* We = (const float*)d_in[11]; const float* be = (const float*)d_in[12];
  const float* Wa = (const float*)d_in[13]; const float* ba = (const float*)d_in[14];
  const float* im = (const float*)d_in[15];
  float* M = (float*)d_out;

  // workspace: [0, 512K) weight pack; then Af (24 MiB); then Bf (24 MiB)
  unsigned* wpack = (unsigned*)d_ws;
  float* Af = (float*)((char*)d_ws + (512u << 10));
  float* Bf = Af + (long)Bb * SSTEPS * MEM;

  prep_weights<<<256, 256, 0, stream>>>(We, Wa, wpack);
  demo_kernel<<<Bb, 256, 0, stream>>>(demo, W1, b1, W2, b2, W3, b3, M);
  init_kernel<<<dim3(1024, Bb), 256, 0, stream>>>((const float4*)im, (float4*)M);
  gates_kernel<<<768, 256, 0, stream>>>(x, mask, vmod, be, ba, wpack, Af, Bf);
  scan_kernel<<<Bb, 256, 0, stream>>>(ids, Af, Bf, M);
}
